// MambaBlock_13795434954874
// MI455X (gfx1250) — matrix operand order repaired and checked by reading, not compiled, not run on hardware
//
#include <hip/hip_runtime.h>
#include <hip/hip_bf16.h>
#include <cstdint>

// ---------------------------------------------------------------------------
// Mamba 2D cross-scan block for gfx1250 (MI455X), wave32 + WMMA f16->f32.
//
// Shapes: B=2, C=64, H=W=64, L=4096, D_INNER=128, D_STATE=16, DT_RANK=4,
// 4 scan directions. GEMMs on the matrix pipe (v_wmma_f32_16x16x32_f16);
// selective scan parallelized as a 3-phase chunked scan (64 chunks of 64)
// with async global->LDS staging (global_load_async_to_lds_b128).
// ---------------------------------------------------------------------------

typedef __attribute__((ext_vector_type(16))) _Float16 v16h;
typedef __attribute__((ext_vector_type(8)))  float    v8f;

#define LSEQ 4096
#define CH   64
#define DI   128
#define NST  16
#define CKLEN  64    // scan chunk length
#define NCHUNK 64    // chunks per sequence

// Direction permutation: sequence index l -> spatial flat index p (row-major h*64+w)
__device__ __forceinline__ int dir_pos(int d, int l) {
    int ll = (d & 1) ? (LSEQ - 1 - l) : l;          // reverse for d1/d3
    if (d & 2) return ((ll & 63) << 6) | (ll >> 6); // transpose for d2/d3
    return ll;
}

// --- CDNA5 async global->LDS copy (ASYNCcnt path) --------------------------
__device__ __forceinline__ void async_g2l_b128(float* lds_dst, const float* gsrc) {
    unsigned loff = (unsigned)(uintptr_t)lds_dst;   // LDS aperture: addr[31:0] = LDS offset
    asm volatile("global_load_async_to_lds_b128 %0, %1, off"
                 :: "v"(loff), "v"(gsrc)
                 : "memory");
}
__device__ __forceinline__ void async_wait_all() {
    asm volatile("s_wait_asynccnt 0" ::: "memory");
}

// --- WMMA fragment loaders (contiguous-k layout, same for A and B, f32 source -> f16 frag) ----
// A (16x32, MxK): lane holds row M = lane&15; K = (lane>>4)*16 + e
__device__ __forceinline__ v16h load_a_row_f32(const float* __restrict__ rowptr, int k0) {
    const int half = (threadIdx.x >> 4) & 1;
    v16h a;
#pragma unroll
    for (int g = 0; g < 2; ++g) {
        const float4* p4 = reinterpret_cast<const float4*>(rowptr + k0 + half * 16 + g * 8);
        float4 f0 = p4[0], f1 = p4[1];
        a[g*8+0] = (_Float16)f0.x; a[g*8+1] = (_Float16)f0.y;
        a[g*8+2] = (_Float16)f0.z; a[g*8+3] = (_Float16)f0.w;
        a[g*8+4] = (_Float16)f1.x; a[g*8+5] = (_Float16)f1.y;
        a[g*8+6] = (_Float16)f1.z; a[g*8+7] = (_Float16)f1.w;
    }
    return a;
}

// B (32x16, KxN): lane holds col N = lane&15; K = (lane>>4)*16 + e
__device__ __forceinline__ v16h load_b_col_f32(const float* __restrict__ colptr, int k0) {
    const int half = (threadIdx.x >> 4) & 1;
    const float4* p4 = reinterpret_cast<const float4*>(colptr + k0 + half * 16);
    v16h b;
#pragma unroll
    for (int q = 0; q < 4; ++q) {
        float4 f = p4[q];
        b[q*4+0] = (_Float16)f.x; b[q*4+1] = (_Float16)f.y;
        b[q*4+2] = (_Float16)f.z; b[q*4+3] = (_Float16)f.w;
    }
    return b;
}

__device__ __forceinline__ v16h load_b_col_f32_guard(const float* __restrict__ colptr,
                                                     int k0, bool valid) {
    v16h b;
    if (valid) {
        b = load_b_col_f32(colptr, k0);
    } else {
#pragma unroll
        for (int e = 0; e < 16; ++e) b[e] = (_Float16)0.0f;
    }
    return b;
}

__device__ __forceinline__ float softplus_f(float x) {
    return (x > 20.f) ? x : __logf(1.0f + __expf(x));
}

// ---------------------------------------------------------------------------
// K1: LayerNorm over C=64. One wave per position (2 channels/lane).
// ---------------------------------------------------------------------------
__global__ void k_layernorm(const float* __restrict__ x, const float* __restrict__ lw,
                            const float* __restrict__ lb, float* __restrict__ xn) {
    int wave = threadIdx.x >> 5;
    int lane = threadIdx.x & 31;
    int pg = blockIdx.x * 8 + wave;        // global position 0..8191
    int b = pg >> 12, p = pg & (LSEQ - 1);
    const float* xb = x + ((size_t)b * CH) * LSEQ + p;
    float v0 = xb[(size_t)lane * LSEQ];
    float v1 = xb[(size_t)(lane + 32) * LSEQ];
    float s  = v0 + v1;
    float ss = v0 * v0 + v1 * v1;
#pragma unroll
    for (int m = 16; m >= 1; m >>= 1) {
        s  += __shfl_xor(s,  m, 32);
        ss += __shfl_xor(ss, m, 32);
    }
    float mean = s * (1.0f / 64.0f);
    float var  = ss * (1.0f / 64.0f) - mean * mean;
    float inv  = rsqrtf(var + 1e-5f);
    float* o = xn + (size_t)pg * CH;
    o[lane]      = (v0 - mean) * inv * lw[lane]      + lb[lane];
    o[lane + 32] = (v1 - mean) * inv * lw[lane + 32] + lb[lane + 32];
}

// ---------------------------------------------------------------------------
// K2: xz = xn_perm @ W_in[d]^T   (4096x64)x(64x256), per (b,d).
// Wave computes one 16x16 tile, K=64 -> 2 WMMA. Direction permutation folded
// into the per-lane A-row gather (no materialized cross-scan).
// ---------------------------------------------------------------------------
__global__ void k_gemm_in(const float* __restrict__ xn, const float* __restrict__ Win,
                          float* __restrict__ xz) {
    int wid  = blockIdx.x * (blockDim.x >> 5) + (threadIdx.x >> 5);
    int lane = threadIdx.x & 31;
    int Nt = wid & 15;
    int Mt = (wid >> 4) & 255;
    int bd = wid >> 12;
    int b = bd >> 2, d = bd & 3;
    int row = lane & 15;
    int lseq = Mt * 16 + row;
    int p = dir_pos(d, lseq);
    const float* arow = xn + (size_t)(b * LSEQ + p) * CH;
    int n = Nt * 16 + row;
    const float* bcol = Win + ((size_t)d * 256 + n) * CH;
    __builtin_prefetch(arow + 32, 0, 1);
    __builtin_prefetch(bcol + 32, 0, 1);
    v8f c = {};
#pragma unroll
    for (int k0 = 0; k0 < CH; k0 += 32) {
        v16h a  = load_a_row_f32(arow, k0);
        v16h bb = load_b_col_f32(bcol, k0);
        c = __builtin_amdgcn_wmma_f32_16x16x32_f16(false, a, false, bb,
                                                   (short)0, c, false, false);
    }
    int half = lane >> 4;
    float* orow = xz + ((size_t)bd * LSEQ + (size_t)Mt * 16) * 256 + Nt * 16 + (lane & 15);
#pragma unroll
    for (int v = 0; v < 8; ++v) {
        int m = v + 8 * half;                 // D layout: M = v + 8*(lane>>4)
        orow[(size_t)m * 256] = c[v];
    }
}

// ---------------------------------------------------------------------------
// K3: depthwise causal conv (k=3, left pad 2) + bias + SiLU on xp channels.
// ---------------------------------------------------------------------------
__global__ void k_conv(const float* __restrict__ xz, const float* __restrict__ cw,
                       const float* __restrict__ cb, float* __restrict__ xp) {
    int t  = blockIdx.x * blockDim.x + threadIdx.x;   // bd*L*DI threads
    int ch = t & (DI - 1);
    int l  = (t >> 7) & (LSEQ - 1);
    int bd = t >> 19;
    int d  = bd & 3;
    const float* w  = cw + ((size_t)d * DI + ch) * 3;
    const float* in = xz + (size_t)bd * LSEQ * 256 + ch;  // xp cols are [0,128)
    float acc = cb[d * DI + ch];
#pragma unroll
    for (int j = 0; j < 3; ++j) {
        int li = l - 2 + j;
        if (li >= 0) acc += w[j] * in[(size_t)li * 256];
    }
    float sx = acc / (1.0f + __expf(-acc));           // SiLU
    xp[((size_t)bd * LSEQ + l) * DI + ch] = sx;
}

// ---------------------------------------------------------------------------
// K4: dbl = xp @ W_xproj[d]^T   (4096x128)x(128x36), N padded to 48.
// ---------------------------------------------------------------------------
__global__ void k_gemm_xproj(const float* __restrict__ xp, const float* __restrict__ Wx,
                             float* __restrict__ dbl) {
    int wid  = blockIdx.x * (blockDim.x >> 5) + (threadIdx.x >> 5);
    int lane = threadIdx.x & 31;
    int Nt = wid % 3;
    int Mt = (wid / 3) & 255;
    int bd = wid / (3 * 256);
    int d  = bd & 3;
    int row = lane & 15;
    const float* arow = xp + ((size_t)bd * LSEQ + (size_t)Mt * 16 + row) * DI;
    int n = Nt * 16 + row;
    bool valid = (n < 36);
    const float* bcol = Wx + ((size_t)d * 36 + (valid ? n : 0)) * DI;
    __builtin_prefetch(arow, 0, 1);
    v8f c = {};
#pragma unroll
    for (int k0 = 0; k0 < DI; k0 += 32) {
        v16h a  = load_a_row_f32(arow, k0);
        v16h bb = load_b_col_f32_guard(bcol, k0, valid);
        c = __builtin_amdgcn_wmma_f32_16x16x32_f16(false, a, false, bb,
                                                   (short)0, c, false, false);
    }
    int half = lane >> 4;
    float* orow = dbl + ((size_t)bd * LSEQ + (size_t)Mt * 16) * 48 + Nt * 16 + (lane & 15);
#pragma unroll
    for (int v = 0; v < 8; ++v) {
        int m = v + 8 * half;
        orow[(size_t)m * 48] = c[v];
    }
}

// ---------------------------------------------------------------------------
// Chunked selective scan (3 phases). Chunk = 64 steps, 64 chunks/sequence.
// Per (ch,state) the decay over a chunk is exp(A * sum(dt)) since A is
// constant, so a chunk summary is just (sum_dt, h_end | h_start=0).
//
// Phase A staging: dbl chunk (64x48) and xp chunk (64x128) are copied to LDS
// with CDNA5 async global->LDS b128 transfers (ASYNCcnt), because every
// thread in the block re-reads the same dt/B/C row each step.
// ---------------------------------------------------------------------------
__device__ __forceinline__ void stage_chunk(const float* __restrict__ dbl,
                                            const float* __restrict__ xp,
                                            int bd, int l0,
                                            float* s_dbl, float* s_xp) {
    const float* gdb = dbl + ((size_t)bd * LSEQ + l0) * 48;   // 64*48 = 3072 floats
    for (int i = threadIdx.x; i < (CKLEN * 48) / 4; i += blockDim.x)
        async_g2l_b128(s_dbl + i * 4, gdb + i * 4);
    const float* gxp = xp + ((size_t)bd * LSEQ + l0) * DI;    // 64*128 = 8192 floats
    for (int i = threadIdx.x; i < (CKLEN * DI) / 4; i += blockDim.x)
        async_g2l_b128(s_xp + i * 4, gxp + i * 4);
    async_wait_all();
    __syncthreads();
}

// Phase A: per-chunk local recurrence (h_start = 0) -> summaries.
__global__ void k_scan_local(const float* __restrict__ xp, const float* __restrict__ dbl,
                             const float* __restrict__ Wdt, const float* __restrict__ bdt,
                             const float* __restrict__ Alog,
                             float* __restrict__ hend, float* __restrict__ dtsum) {
    __shared__ float s_dbl[CKLEN * 48];
    __shared__ float s_xp[CKLEN * DI];
    int blk = blockIdx.x;                 // 0..511 = (bd, chunk)
    int bd = blk >> 6, ck = blk & (NCHUNK - 1);
    int d = bd & 3;
    int l0 = ck * CKLEN;
    stage_chunk(dbl, xp, bd, l0, s_dbl, s_xp);

    int t = threadIdx.x, ch = t >> 2, sg = t & 3;   // 4 states / thread
    float A[4], wdt[4];
#pragma unroll
    for (int j = 0; j < 4; ++j) {
        A[j]   = -__expf(Alog[((size_t)d * DI + ch) * NST + sg * 4 + j]);
        wdt[j] = Wdt[((size_t)d * DI + ch) * 4 + j];
    }
    float bdt_c = bdt[d * DI + ch];
    float h[4] = {0.f, 0.f, 0.f, 0.f};
    float dts = 0.f;
    for (int l = 0; l < CKLEN; ++l) {
        const float* dl = s_dbl + l * 48;
        float dt = softplus_f(dl[0]*wdt[0] + dl[1]*wdt[1] + dl[2]*wdt[2] + dl[3]*wdt[3] + bdt_c);
        dts += dt;
        float du = dt * s_xp[l * DI + ch];
        const float* Bl = dl + 4 + sg * 4;
#pragma unroll
        for (int j = 0; j < 4; ++j)
            h[j] = __expf(dt * A[j]) * h[j] + du * Bl[j];
    }
    size_t base = ((size_t)blk * DI + ch) * NST + sg * 4;
#pragma unroll
    for (int j = 0; j < 4; ++j) hend[base + j] = h[j];
    if (sg == 0) dtsum[(size_t)blk * DI + ch] = dts;
}

// Phase B: serial prefix over the 64 chunk summaries (8 blocks, one per seq).
__global__ void k_scan_combine(const float* __restrict__ hend, const float* __restrict__ dtsum,
                               const float* __restrict__ Alog, float* __restrict__ hstart) {
    int bd = blockIdx.x, d = bd & 3;
    int t = threadIdx.x, ch = t >> 2, sg = t & 3;
    float A[4];
#pragma unroll
    for (int j = 0; j < 4; ++j)
        A[j] = -__expf(Alog[((size_t)d * DI + ch) * NST + sg * 4 + j]);
    float hs[4] = {0.f, 0.f, 0.f, 0.f};
    for (int ck = 0; ck < NCHUNK; ++ck) {
        int blk = bd * NCHUNK + ck;
        size_t base = ((size_t)blk * DI + ch) * NST + sg * 4;
        float dts = dtsum[(size_t)blk * DI + ch];
#pragma unroll
        for (int j = 0; j < 4; ++j) {
            hstart[base + j] = hs[j];
            hs[j] = __expf(A[j] * dts) * hs[j] + hend[base + j];
        }
    }
}

// Phase C: re-run local recurrence seeded with chunk-start state; emit
// y = (h.C + xp*D) * SiLU(z). 4-lane shuffle reduction for the state sum.
__global__ void k_scan_apply(const float* __restrict__ xz, const float* __restrict__ xp,
                             const float* __restrict__ dbl, const float* __restrict__ Wdt,
                             const float* __restrict__ bdt, const float* __restrict__ Alog,
                             const float* __restrict__ Dpar, const float* __restrict__ hstart,
                             float* __restrict__ ybuf) {
    __shared__ float s_dbl[CKLEN * 48];
    __shared__ float s_xp[CKLEN * DI];
    int blk = blockIdx.x;
    int bd = blk >> 6, ck = blk & (NCHUNK - 1);
    int d = bd & 3;
    int l0 = ck * CKLEN;
    stage_chunk(dbl, xp, bd, l0, s_dbl, s_xp);

    int t = threadIdx.x, ch = t >> 2, sg = t & 3;
    float A[4], wdt[4], h[4];
    size_t base = ((size_t)blk * DI + ch) * NST + sg * 4;
#pragma unroll
    for (int j = 0; j < 4; ++j) {
        A[j]   = -__expf(Alog[((size_t)d * DI + ch) * NST + sg * 4 + j]);
        wdt[j] = Wdt[((size_t)d * DI + ch) * 4 + j];
        h[j]   = hstart[base + j];
    }
    float bdt_c = bdt[d * DI + ch];
    float Dc    = Dpar[d * DI + ch];
    const float* zb = xz + ((size_t)bd * LSEQ + l0) * 256 + DI + ch;  // z = cols [128,256)
    float* yb = ybuf + ((size_t)bd * LSEQ + l0) * DI + ch;
    for (int l = 0; l < CKLEN; ++l) {
        const float* dl = s_dbl + l * 48;
        float dt = softplus_f(dl[0]*wdt[0] + dl[1]*wdt[1] + dl[2]*wdt[2] + dl[3]*wdt[3] + bdt_c);
        float xv = s_xp[l * DI + ch];
        float du = dt * xv;
        const float* Bl = dl + 4  + sg * 4;
        const float* Cl = dl + 20 + sg * 4;
        float acc = 0.f;
#pragma unroll
        for (int j = 0; j < 4; ++j) {
            h[j] = __expf(dt * A[j]) * h[j] + du * Bl[j];
            acc += h[j] * Cl[j];
        }
        acc += __shfl_xor(acc, 1, 32);
        acc += __shfl_xor(acc, 2, 32);
        if (sg == 0) {
            float zv = zb[(size_t)l * 256];
            float y  = (acc + xv * Dc) * (zv / (1.0f + __expf(-zv)));
            yb[(size_t)l * DI] = y;
        }
    }
}

// ---------------------------------------------------------------------------
// K6: out = sum_d y_d[l_d(p)] @ W_out[d]^T, combined in one accumulator
// (16 WMMAs / tile). Direction gather folded into A-row index; stores land
// directly in (B,C,H,W) layout (8 contiguous floats per lane).
// ---------------------------------------------------------------------------
__global__ void k_gemm_out(const float* __restrict__ ybuf, const float* __restrict__ Wout,
                           float* __restrict__ out) {
    int wid  = blockIdx.x * (blockDim.x >> 5) + (threadIdx.x >> 5);
    int lane = threadIdx.x & 31;
    int Nt = wid & 3;              // 64 output channels -> 4 N tiles
    int Mt = wid >> 2;             // 0..511 over B*L rows
    int row = lane & 15;
    int pg = Mt * 16 + row;
    int b = pg >> 12;
    int p = pg & (LSEQ - 1);
    int n = Nt * 16 + row;
    int pt = ((p & 63) << 6) | (p >> 6);   // transposed spatial index
    v8f c = {};
#pragma unroll
    for (int d = 0; d < 4; ++d) {
        int l = (d == 0) ? p
              : (d == 1) ? (LSEQ - 1 - p)
              : (d == 2) ? pt
                         : (LSEQ - 1 - pt);
        const float* arow = ybuf + ((size_t)(b * 4 + d) * LSEQ + l) * DI;
        const float* bcol = Wout + ((size_t)d * CH + n) * DI;
        __builtin_prefetch(arow, 0, 1);
#pragma unroll
        for (int k0 = 0; k0 < DI; k0 += 32) {
            v16h a  = load_a_row_f32(arow, k0);
            v16h bb = load_b_col_f32(bcol, k0);
            c = __builtin_amdgcn_wmma_f32_16x16x32_f16(false, a, false, bb,
                                                       (short)0, c, false, false);
        }
    }
    int half = lane >> 4;
    int cidx = Nt * 16 + (lane & 15);
    int p0 = (Mt & 255) * 16 + half * 8;   // D rows are contiguous spatial positions
    float* o = out + (size_t)(b * CH + cidx) * LSEQ + p0;
#pragma unroll
    for (int v = 0; v < 8; ++v) o[v] = c[v];
}

// ---------------------------------------------------------------------------
extern "C" void kernel_launch(void* const* d_in, const int* in_sizes, int n_in,
                              void* d_out, int out_size, void* d_ws, size_t ws_size,
                              hipStream_t stream) {
    const float* x    = (const float*)d_in[0];
    const float* lnw  = (const float*)d_in[1];
    const float* lnb  = (const float*)d_in[2];
    const float* Win  = (const float*)d_in[3];
    const float* cw   = (const float*)d_in[4];
    const float* cb   = (const float*)d_in[5];
    const float* Wx   = (const float*)d_in[6];
    const float* Wdt  = (const float*)d_in[7];
    const float* bdt  = (const float*)d_in[8];
    const float* Alog = (const float*)d_in[9];
    const float* Dpar = (const float*)d_in[10];
    const float* Wout = (const float*)d_in[11];
    float* out = (float*)d_out;

    // Workspace layout (fp32), ~84 MB total -> fully L2-resident (192 MB).
    float* xn   = (float*)d_ws;                        // [2][4096][64]
    float* xz   = xn   + (size_t)2 * LSEQ * CH;        // [8][4096][256] (xp_pre | z)
    float* xp   = xz   + (size_t)8 * LSEQ * 256;       // [8][4096][128]
    float* dbl  = xp   + (size_t)8 * LSEQ * DI;        // [8][4096][48]  (dt4|B16|C16|pad)
    float* yb   = dbl  + (size_t)8 * LSEQ * 48;        // [8][4096][128]
    float* hend = yb   + (size_t)8 * LSEQ * DI;        // [8*64][128][16]
    float* hst  = hend + (size_t)8 * NCHUNK * DI * NST;// [8*64][128][16]
    float* dts  = hst  + (size_t)8 * NCHUNK * DI * NST;// [8*64][128]

    k_layernorm  <<<1024,  256, 0, stream>>>(x, lnw, lnb, xn);
    k_gemm_in    <<<4096,  256, 0, stream>>>(xn, Win, xz);    // 8*256*16 wave-tiles
    k_conv       <<<16384, 256, 0, stream>>>(xz, cw, cb, xp);
    k_gemm_xproj <<<768,   256, 0, stream>>>(xp, Wx, dbl);    // 8*256*3 wave-tiles
    k_scan_local <<<512,   512, 0, stream>>>(xp, dbl, Wdt, bdt, Alog, hend, dts);
    k_scan_combine<<<8,    512, 0, stream>>>(hend, dts, Alog, hst);
    k_scan_apply <<<512,   512, 0, stream>>>(xz, xp, dbl, Wdt, bdt, Alog, Dpar, hst, yb);
    k_gemm_out   <<<256,   256, 0, stream>>>(yb, Wout, out);  // 512*4 wave-tiles
}
